// GroupedQueryAttention_17008070492379
// MI455X (gfx1250) — matrix operand order repaired and checked by reading, not compiled, not run on hardware
//
#include <hip/hip_runtime.h>
#include <stdint.h>

// ---------------- problem constants ----------------
#define D_IN   2048
#define D_OUTD 2048
#define NHEADS 16
#define NKV    4
#define HDIM   128
#define GRP    4
#define BB     2
#define NN     2048   // query length == kv length

// ---------------- WMMA vector types ----------------
typedef __attribute__((ext_vector_type(16))) __bf16 v16bf;
typedef __attribute__((ext_vector_type(8)))  float  v8f;

__device__ __forceinline__ uint16_t f2bf(float f) {
  union { float f; uint32_t u; } x; x.f = f;
  uint32_t r = x.u + 0x7FFFu + ((x.u >> 16) & 1u);  // round-to-nearest-even
  return (uint16_t)(r >> 16);
}
__device__ __forceinline__ uint32_t pkbf(float lo, float hi) {
  return (uint32_t)f2bf(lo) | ((uint32_t)f2bf(hi) << 16);
}

// ---- CDNA5 async global->LDS DMA, GVS addressing (SGPR base + 32-bit VGPR
// offset); the 24-bit inst offset advances BOTH the global src and LDS dst,
// so one base pair + 4 immediates copies a contiguous 64B row with no per-
// chunk address math. Tracked by ASYNCcnt. ----
__device__ __forceinline__ void async_row64(uint32_t lds, uint32_t goff,
                                            const void* base) {
  asm volatile(
      "global_load_async_to_lds_b128 %0, %1, %2\n\t"
      "global_load_async_to_lds_b128 %0, %1, %2 offset:16\n\t"
      "global_load_async_to_lds_b128 %0, %1, %2 offset:32\n\t"
      "global_load_async_to_lds_b128 %0, %1, %2 offset:48"
      :: "v"(lds), "v"(goff), "s"((uint64_t)(uintptr_t)base)
      : "memory");
}
__device__ __forceinline__ void wait_async0() {
#if __has_builtin(__builtin_amdgcn_s_wait_asynccnt)
  __builtin_amdgcn_s_wait_asynccnt(0);
#else
  asm volatile("s_wait_asynccnt 0x0" ::: "memory");
#endif
}

// =====================================================================
// Generic 128x128 block-tile bf16 WMMA GEMM core, double-buffered LDS.
//   C[M,N] = A[M,K] * B[K,N]
// ATY/BTY/CTY: 0 = float32 in memory, 1 = bf16 (uint16 storage)
// BTRANS: B given as [N][K] row-major (multiply by B_src^T); K runs are
//         contiguous in B_src rows -> async-DMA friendly.
// Block = 128 threads = 4 waves (wave32); each wave computes 64x64.
// All tile dims divide M/N/K for every call in this problem.
// =====================================================================
template<int ATY, int BTY, bool BTRANS, int CTY>
__device__ __forceinline__ void gemm_core(
    const void* __restrict__ Av, int lda,
    const void* __restrict__ Bv, int ldb,
    void*       __restrict__ Cv, int ldc,
    int Kdim, int m0, int n0)
{
  __shared__ uint16_t Asub[2][128][40];   // [buf][m][k], +8 pad (80B rows, 16B aligned)
  __shared__ uint16_t Bsub[2][128][40];   // [buf][n][k]

  const int t    = threadIdx.x;        // 0..127
  const int lane = t & 31;
  const int wave = t >> 5;
  const int wm   = (wave >> 1) * 64;
  const int wn   = (wave & 1) * 64;
  const int mr   = lane & 15;
  const int hi   = lane >> 4;

  constexpr bool ANY_ASYNC = (ATY == 1) || BTRANS;

  // ---- tile staging: global -> LDS buf (async DMA for bf16, convert for f32)
  auto stage = [&](int buf, int k0) {
    // A tile: rows m0..m0+127, cols k0..k0+31
    if (ATY == 0) {
      const float4* src = (const float4*)((const float*)Av + (size_t)(m0 + t) * lda + k0);
      #pragma unroll
      for (int j = 0; j < 8; j++) {
        float4 v = src[j];
        uint2 q; q.x = pkbf(v.x, v.y); q.y = pkbf(v.z, v.w);
        *(uint2*)&Asub[buf][t][4*j] = q;
      }
    } else {
      const uint32_t goff = (uint32_t)(((size_t)(m0 + t) * lda + k0) * 2u);
      async_row64((uint32_t)(uintptr_t)&Asub[buf][t][0], goff, Av);
    }
    // B tile into Bsub[n][k]
    if (BTRANS) {
      const uint32_t goff = (uint32_t)(((size_t)(n0 + t) * ldb + k0) * 2u);
      async_row64((uint32_t)(uintptr_t)&Bsub[buf][t][0], goff, Bv);
    } else if (BTY == 0) {
      const float* src = (const float*)Bv;
      #pragma unroll
      for (int k = 0; k < 32; k += 2) {      // coalesced across threads per k-row
        float a = src[(size_t)(k0 + k)     * ldb + n0 + t];
        float b = src[(size_t)(k0 + k + 1) * ldb + n0 + t];
        *(uint32_t*)&Bsub[buf][t][k] = pkbf(a, b);
      }
    } else {
      const uint16_t* src = (const uint16_t*)Bv;
      #pragma unroll
      for (int k = 0; k < 32; k += 2) {
        uint32_t lo = src[(size_t)(k0 + k)     * ldb + n0 + t];
        uint32_t h2 = src[(size_t)(k0 + k + 1) * ldb + n0 + t];
        *(uint32_t*)&Bsub[buf][t][k] = lo | (h2 << 16);
      }
    }
  };

  v8f acc[4][4];
  #pragma unroll
  for (int i = 0; i < 4; i++)
    #pragma unroll
    for (int j = 0; j < 4; j++)
      #pragma unroll
      for (int r = 0; r < 8; r++)
        acc[i][j][r] = 0.f;

  stage(0, 0);
  if (ANY_ASYNC) wait_async0();
  __syncthreads();

  for (int k0 = 0; k0 < Kdim; k0 += 32) {
    const int cur = (k0 >> 5) & 1;
    if (k0 + 32 < Kdim) stage(cur ^ 1, k0 + 32);
    if (ATY == 0 && k0 + 64 < Kdim)
      __builtin_prefetch((const float*)Av + (size_t)(m0 + t) * lda + k0 + 64, 0, 3);

    // ---------- fragments (ISA 16-bit A 16x32 / B layouts) ----------
    union U { uint4 u[2]; v16bf v; };
    v16bf af[4], bfr[4];
    #pragma unroll
    for (int i = 0; i < 4; i++) {
      U u;
      const int m = wm + i*16 + mr;
      u.u[0] = *(const uint4*)&Asub[cur][m][hi*8];        // K = hi*8 .. +8
      u.u[1] = *(const uint4*)&Asub[cur][m][16 + hi*8];   // K = 16+hi*8 .. +8
      af[i] = u.v;
    }
    #pragma unroll
    for (int j = 0; j < 4; j++) {
      U u;
      const int n = wn + j*16 + mr;
      u.u[0] = *(const uint4*)&Bsub[cur][n][hi*8];        // K = hi*8 .. +8
      u.u[1] = *(const uint4*)&Bsub[cur][n][16 + hi*8];
      bfr[j] = u.v;
    }

    #pragma unroll
    for (int i = 0; i < 4; i++)
      #pragma unroll
      for (int j = 0; j < 4; j++)
        acc[i][j] = __builtin_amdgcn_wmma_f32_16x16x32_bf16(
            false, af[i], false, bfr[j], (short)0, acc[i][j], false, false);

    if (ANY_ASYNC) wait_async0();   // own DMA writes to buf^1 landed
    __syncthreads();                // all waves: writes visible, reads done
  }

  // ---------- store C (C/D layout: VGPR r -> M = r + hi*8) ----------
  #pragma unroll
  for (int i = 0; i < 4; i++) {
    #pragma unroll
    for (int j = 0; j < 4; j++) {
      const int n = n0 + wn + j*16 + mr;
      #pragma unroll
      for (int r = 0; r < 8; r++) {
        const int m = m0 + wm + i*16 + hi*8 + r;
        float v = acc[i][j][r];
        if (CTY == 0) ((float*)Cv)[(size_t)m * ldc + n] = v;
        else          ((uint16_t*)Cv)[(size_t)m * ldc + n] = f2bf(v);
      }
    }
  }
}

// ---------------- plain GEMM wrapper ----------------
template<int ATY, int BTY, bool BTRANS, int CTY>
__global__ __launch_bounds__(128)
void gemm_kernel(const void* __restrict__ A, int lda,
                 const void* __restrict__ B, int ldb,
                 void* __restrict__ C, int ldc, int Kdim)
{
  gemm_core<ATY, BTY, BTRANS, CTY>(A, lda, B, ldb, C, ldc, Kdim,
                                   blockIdx.y * 128, blockIdx.x * 128);
}

// ---------------- scores: W[z] = Q_head @ K_head^T (raw, no 1/sqrt(d)) ----
__global__ __launch_bounds__(128)
void scores_kernel(const uint16_t* __restrict__ Qbf,
                   const uint16_t* __restrict__ Kbf,
                   float* __restrict__ Wt)
{
  const int z   = blockIdx.z;          // weights batch: b*16 + g*4 + hkv
  const int b   = z >> 4;
  const int r   = z & 15;
  const int g   = r >> 2;
  const int hkv = r & 3;
  const int hh  = hkv * GRP + g;       // query-head index in (h g) packing
  const uint16_t* A  = Qbf + (size_t)b * NN * D_OUTD + (size_t)hh * HDIM;
  const uint16_t* Bm = Kbf + (size_t)b * NN * (NKV * HDIM) + (size_t)hkv * HDIM;
  float* C = Wt + (size_t)z * NN * NN;
  gemm_core<1, 1, true, 0>(A, D_OUTD, Bm, NKV * HDIM, C, NN, HDIM,
                           blockIdx.y * 128, blockIdx.x * 128);
}

// ---------------- ctx: Ctx[:, head*128:+128] = softmax(W[z]) @ V_head -----
__global__ __launch_bounds__(128)
void ctx_kernel(const float* __restrict__ Wt,
                const uint16_t* __restrict__ Vbf,
                uint16_t* __restrict__ Ctx)
{
  const int z   = blockIdx.z;
  const int b   = z >> 4;
  const int r   = z & 15;              // = g*4 + hkv = output column block
  const int hkv = r & 3;
  const float*    A  = Wt  + (size_t)z * NN * NN;
  const uint16_t* Bm = Vbf + (size_t)b * NN * (NKV * HDIM) + (size_t)hkv * HDIM;
  uint16_t* C = Ctx + (size_t)b * NN * D_OUTD + (size_t)r * HDIM;
  gemm_core<0, 1, false, 1>(A, NN, Bm, NKV * HDIM, C, D_OUTD, NN,
                            blockIdx.y * 128, blockIdx.x * 128);
}

// ---------------- in-place row softmax over S=2048 ----------------
__global__ __launch_bounds__(256)
void softmax_kernel(float* __restrict__ Wt)
{
  float* row = Wt + (size_t)blockIdx.x * NN;
  const int t  = threadIdx.x;
  const int wv = t >> 5, ln = t & 31;
  __shared__ float red[8];

  float v[8];
  float m = -3.402823466e38f;
  #pragma unroll
  for (int i = 0; i < 8; i++) { v[i] = row[t + i * 256]; m = fmaxf(m, v[i]); }
  #pragma unroll
  for (int off = 16; off > 0; off >>= 1) m = fmaxf(m, __shfl_xor(m, off, 32));
  if (ln == 0) red[wv] = m;
  __syncthreads();
  #pragma unroll
  for (int i = 0; i < 8; i++) m = fmaxf(m, red[i]);

  float s = 0.f;
  #pragma unroll
  for (int i = 0; i < 8; i++) { v[i] = __expf(v[i] - m); s += v[i]; }
  #pragma unroll
  for (int off = 16; off > 0; off >>= 1) s += __shfl_xor(s, off, 32);
  __syncthreads();
  if (ln == 0) red[wv] = s;
  __syncthreads();
  s = 0.f;
  #pragma unroll
  for (int i = 0; i < 8; i++) s += red[i];
  const float inv = 1.f / s;
  #pragma unroll
  for (int i = 0; i < 8; i++) row[t + i * 256] = v[i] * inv;
}

// =====================================================================
extern "C" void kernel_launch(void* const* d_in, const int* in_sizes, int n_in,
                              void* d_out, int out_size, void* d_ws, size_t ws_size,
                              hipStream_t stream)
{
  (void)in_sizes; (void)n_in; (void)out_size; (void)ws_size;
  const float* query = (const float*)d_in[0];
  const float* key   = (const float*)d_in[1];
  const float* value = (const float*)d_in[2];
  const float* Wq    = (const float*)d_in[3];
  const float* Wk    = (const float*)d_in[4];
  const float* Wv    = (const float*)d_in[5];
  const float* Wo    = (const float*)d_in[6];

  float* out     = (float*)d_out;                     // (2, 2048, 2048)
  float* weights = out + (size_t)BB * NN * D_OUTD;    // (32, 2048, 2048)

  // workspace layout (bf16 buffers); Ctx reuses the Qbf region (dead then)
  char* ws = (char*)d_ws;
  uint16_t* Qbf = (uint16_t*)(ws);                               // 16 MB
  uint16_t* Kbf = (uint16_t*)(ws + (size_t)16 * 1024 * 1024);    //  4 MB
  uint16_t* Vbf = (uint16_t*)(ws + (size_t)20 * 1024 * 1024);    //  4 MB
  uint16_t* Ctx = Qbf;                                           // reuse

  const dim3 blk(128);
  const int MROWS = BB * NN;                                     // 4096

  // Q/K/V projections: fp32 -> bf16 WMMA -> bf16 activations
  gemm_kernel<0,0,false,1><<<dim3(D_OUTD/128, MROWS/128), blk, 0, stream>>>(
      query, D_IN, Wq, D_OUTD, Qbf, D_OUTD, D_IN);
  gemm_kernel<0,0,false,1><<<dim3((NKV*HDIM)/128, MROWS/128), blk, 0, stream>>>(
      key, D_IN, Wk, NKV*HDIM, Kbf, NKV*HDIM, D_IN);
  gemm_kernel<0,0,false,1><<<dim3((NKV*HDIM)/128, MROWS/128), blk, 0, stream>>>(
      value, D_IN, Wv, NKV*HDIM, Vbf, NKV*HDIM, D_IN);

  // raw attention scores straight into the weights output region (async DMA staging)
  scores_kernel<<<dim3(NN/128, NN/128, BB*NHEADS), blk, 0, stream>>>(Qbf, Kbf, weights);

  // in-place softmax over last dim
  softmax_kernel<<<dim3(BB*NHEADS*NN), dim3(256), 0, stream>>>(weights);

  // ctx = weights @ V  (weights fp32 -> bf16 fused in staging)
  ctx_kernel<<<dim3(1, NN/128, BB*NHEADS), blk, 0, stream>>>(weights, Vbf, Ctx);

  // output projection: ctx_bf16 @ Wo -> fp32
  gemm_kernel<1,0,false,0><<<dim3(D_OUTD/128, MROWS/128), blk, 0, stream>>>(
      Ctx, D_OUTD, Wo, D_OUTD, out, D_OUTD, D_IN);
}
